// CudaW4A16Linear_46686294507979
// MI455X (gfx1250) — compile-verified, matrix operand order checked by reading, not run on hardware
//
#include <hip/hip_runtime.h>
#include <hip/hip_bf16.h>

// ---------------------------------------------------------------------------
// W4A16 grouped-quant linear:  out[8,11008] = x[8,4096] @ dequant(W)[4096,11008]
//   qweight: [11008, 512] i32  (8 int4 nibbles per dword, nibble i -> k = 8d+i)
//   qzeros : [11008, 4]   i32  (8 int4 zero-points per dword, one per group)
//   scales : [11008, 32]  f32  (group size 128)
// HBM-bound on the 22.5MB weight stream. Inner loop feeds WMMA the raw biased
// f16 values (1024+q), built purely with v_perm_b32 (no math ops), and the
// affine dequant  s*(q-z) = s*(1024+q) - s*(1024+z)  is applied once per
// 128-K group via f32 FMAs using per-group activation row-sums computed
// during LDS staging. v_wmma_f32_16x16x32_f16, M=8 padded to 16 zero rows,
// K split 8-way with f32 global atomics into a pre-zeroed output.
// ---------------------------------------------------------------------------

#define IN_FEATURES   4096
#define OUT_FEATURES  11008
#define GROUP_SIZE    128
#define GROUPS        (IN_FEATURES / GROUP_SIZE)       // 32
#define KD8           (IN_FEATURES / 8)                // 512 dwords per row
#define KSPLIT        8
#define KCHUNK        (IN_FEATURES / KSPLIT)           // 512
#define GROUPS_PER_CHUNK (KCHUNK / GROUP_SIZE)         // 4
#define COLS_PER_BLOCK 128                             // 8 waves * 16 cols
#define NBLK_X        (OUT_FEATURES / COLS_PER_BLOCK)  // 86

typedef __attribute__((ext_vector_type(16))) _Float16 v16h;
typedef __attribute__((ext_vector_type(8)))  _Float16 v8h;
typedef __attribute__((ext_vector_type(8)))  float    v8f;
typedef __attribute__((ext_vector_type(4)))  float    v4f;

__global__ __launch_bounds__(256) void w4a16_zero_kernel(float* __restrict__ out, int n)
{
    int i = blockIdx.x * blockDim.x + threadIdx.x;
    if (i < n) out[i] = 0.0f;
}

// Expand one weight dword (8 int4) into 4 dwords of packed f16 {1024+q_even,
// 1024+q_odd} using only v_perm_b32: isolate nibbles, interleave, and inject
// the 0x64 exponent byte straight from a 0x64646464 source register.
__device__ __forceinline__ void nib2h16(unsigned w, unsigned* dst)
{
    const unsigned k64 = 0x64646464u;
    unsigned lo  = w & 0x0F0F0F0Fu;            // n0,n2,n4,n6
    unsigned hi4 = (w >> 4) & 0x0F0F0F0Fu;     // n1,n3,n5,n7
    unsigned e0 = __builtin_amdgcn_perm(hi4, lo, 0x05010400u);  // bytes n0,n1,n2,n3
    unsigned e1 = __builtin_amdgcn_perm(hi4, lo, 0x07030602u);  // bytes n4,n5,n6,n7
    dst[0] = __builtin_amdgcn_perm(k64, e0, 0x04010400u);  // {0x6400|n0, 0x6400|n1}
    dst[1] = __builtin_amdgcn_perm(k64, e0, 0x04030402u);  // {0x6400|n2, 0x6400|n3}
    dst[2] = __builtin_amdgcn_perm(k64, e1, 0x04010400u);  // {0x6400|n4, 0x6400|n5}
    dst[3] = __builtin_amdgcn_perm(k64, e1, 0x04030402u);  // {0x6400|n6, 0x6400|n7}
}

__global__ __launch_bounds__(256) void w4a16_wmma_kernel(
    const float* __restrict__ x,          // [8, 4096]
    const int*   __restrict__ qweight,    // [11008, 512]
    const int*   __restrict__ qzeros,     // [11008, 4]
    const float* __restrict__ scales,     // [11008, 32]
    float*       __restrict__ out)        // [8, 11008]
{
    // 16 rows x 512 K of f16 activations; rows 8..15 zero so WMMA A-fragment
    // loads are uniform (EXEC all-ones around WMMA as the ISA requires).
    __shared__ __align__(32) _Float16 xs[16 * KCHUNK];              // 16 KB
    __shared__ __align__(16) float    rsum[GROUPS_PER_CHUNK * 8];   // [g][row]

    const int t  = threadIdx.x;
    const int kc = blockIdx.y;            // K-chunk id, 0..7

    if (t < GROUPS_PER_CHUNK * 8) rsum[t] = 0.0f;
    __syncthreads();

    // ---- Stage x chunk into LDS as f16 + per-group row sums ----
    {
        int base = t * 16;                // 16 consecutive halves, same row+group
        int row  = base >> 9;             // /512
        int k    = base & (KCHUNK - 1);
        const float* src = x + row * IN_FEATURES + kc * KCHUNK + k;
        float lsum = 0.0f;
        #pragma unroll
        for (int i = 0; i < 16; ++i) {
            _Float16 h = (_Float16)src[i];
            xs[row * KCHUNK + k + i] = h;
            lsum += (float)h;             // sum of f16-rounded activations
        }
        int g = k >> 7;                   // group within chunk (k/128)
        atomicAdd(&rsum[g * 8 + row], lsum);
        // zero rows 8..15
        int zb = 8 * KCHUNK + t * 16;
        #pragma unroll
        for (int i = 0; i < 16; ++i)
            xs[zb + i] = (_Float16)0.0f;
    }
    __syncthreads();

    const int wave = t >> 5;              // 0..7
    const int lane = t & 31;
    const int m    = lane & 15;           // A row (M) / B column (N) selector
    const int hi   = lane >> 4;           // half-wave selects K octet
    const int col  = blockIdx.x * COLS_PER_BLOCK + wave * 16 + m;

    const unsigned* qw_row = (const unsigned*)(qweight) + (size_t)col * KD8;

    // Per-chunk group metadata: 4 scales (one b128) + 4 zero nibbles (one b32).
    v4f s4 = *(const v4f*)(scales + (size_t)col * GROUPS + kc * GROUPS_PER_CHUNK);
    const unsigned zdw = ((const unsigned*)qzeros)[(size_t)col * 4 + (kc >> 1)];

    v8f c = {};

    #pragma unroll
    for (int g4 = 0; g4 < GROUPS_PER_CHUNK; ++g4) {
        // ---- accumulate raw biased products for this 128-K group ----
        v8f cg = {};
        #pragma unroll
        for (int it = 0; it < 4; ++it) {
            const int krel = g4 * GROUP_SIZE + it * 32;        // K within chunk
            const int d0   = ((kc * KCHUNK + krel) >> 3) + hi; // dword index in row
            const unsigned w0 = qw_row[d0];                    // K = base + 8*hi + 0..7
            const unsigned w1 = qw_row[d0 + 2];                // K = base + 16 + 8*hi + 0..7

            union { v16h v; unsigned u[8]; } B;                // B frag: 1024+q
            nib2h16(w0, &B.u[0]);
            nib2h16(w1, &B.u[4]);

            union { v16h v; v8h h8[2]; } A;                    // A frag (16x32)
            const _Float16* arow = &xs[m * KCHUNK + krel + hi * 8];
            A.h8[0] = *(const v8h*)(arow);        // K = krel + 8*hi + 0..7
            A.h8[1] = *(const v8h*)(arow + 16);   // K = krel + 16 + 8*hi + 0..7

            cg = __builtin_amdgcn_wmma_f32_16x16x32_f16(
                     false, A.v, false, B.v, (short)0, cg, false, false);
        }

        // ---- per-group affine fix-up: c += s*cg - s*(1024+z)*rowsum ----
        const float s  = s4[g4];
        const int   z  = (zdw >> (((kc * GROUPS_PER_CHUNK + g4) & 7) * 4)) & 15;
        const float tt = -(1024.0f + (float)z) * s;
        const v4f* rp = (const v4f*)&rsum[g4 * 8];
        const v4f r0 = rp[0];
        const v4f r1 = rp[1];
        #pragma unroll
        for (int r = 0; r < 4; ++r)
            c[r] = __builtin_fmaf(s, cg[r], __builtin_fmaf(tt, r0[r], c[r]));
        #pragma unroll
        for (int r = 4; r < 8; ++r)
            c[r] = __builtin_fmaf(s, cg[r], __builtin_fmaf(tt, r1[r - 4], c[r]));
    }

    // C layout: VGPR r, lanes 0-15 -> (M=r, N=lane); rows 8..15 (and the
    // correction applied to them) are padding, never stored.
    if (lane < 16) {
        #pragma unroll
        for (int r = 0; r < 8; ++r) {
            float* p = out + (size_t)r * OUT_FEATURES + col;
            __hip_atomic_fetch_add(p, c[r], __ATOMIC_RELAXED, __HIP_MEMORY_SCOPE_AGENT);
        }
    }
}

extern "C" void kernel_launch(void* const* d_in, const int* in_sizes, int n_in,
                              void* d_out, int out_size, void* d_ws, size_t ws_size,
                              hipStream_t stream)
{
    const float* x        = (const float*)d_in[0];
    const int*   qweight  = (const int*)d_in[1];
    const int*   qzeros   = (const int*)d_in[2];
    const float* scales   = (const float*)d_in[3];
    float*       out      = (float*)d_out;

    const int n = 8 * OUT_FEATURES;
    w4a16_zero_kernel<<<(n + 255) / 256, 256, 0, stream>>>(out, n);

    dim3 grid(NBLK_X, KSPLIT);
    w4a16_wmma_kernel<<<grid, 256, 0, stream>>>(x, qweight, qzeros, scales, out);
}
